// TransformerLayer_5841155523188
// MI455X (gfx1250) — compile-verified, matrix operand order checked by reading, not run on hardware
//
#include <hip/hip_runtime.h>
#include <math.h>
#include <stdint.h>

// ---------------- types ----------------
typedef unsigned short u16;
typedef __attribute__((ext_vector_type(16))) __bf16 v16bf;
typedef __attribute__((ext_vector_type(8)))  float  v8f;

// Use CDNA5 async global->LDS copies (ASYNCcnt) via inline asm.
#define USE_ASYNC_LDS 1

template <int N>
__device__ __forceinline__ void wait_async() {
#if __has_builtin(__builtin_amdgcn_s_wait_asynccnt)
    __builtin_amdgcn_s_wait_asynccnt(N);
#else
    asm volatile("s_wait_asynccnt %0" :: "i"(N) : "memory");
#endif
}

// GLOBAL_LOAD_ASYNC_TO_LDS_B128: vdst = LDS byte address (u32 VGPR),
// vaddr = 64-bit global address (VGPR pair), GV mode (saddr = off).
__device__ __forceinline__ void async_copy_b128(const void* gsrc, void* ldst) {
    unsigned ldsa = (unsigned)(uintptr_t)ldst;                 // addr[31:0] = LDS offset
    unsigned long long ga = (unsigned long long)(uintptr_t)gsrc;
    asm volatile("global_load_async_to_lds_b128 %0, %1, off"
                 :: "v"(ldsa), "v"(ga) : "memory");
}

__device__ __forceinline__ u16 f2bf(float f) {
    unsigned u = __float_as_uint(f);
    u += 0x7fffu + ((u >> 16) & 1u);   // round-to-nearest-even
    return (u16)(u >> 16);
}

// ---------------- fp32 -> bf16 cast ----------------
__global__ __launch_bounds__(256) void cast_f32_bf16(const float* __restrict__ in,
                                                     u16* __restrict__ out, int n) {
    int i = blockIdx.x * blockDim.x + threadIdx.x;
    int stride = gridDim.x * blockDim.x;
    for (; i < n; i += stride) out[i] = f2bf(in[i]);
}

// ---------------- WMMA GEMM core ----------------
// C[m,n] = alpha * (sum_k A[m,k]*B[n,k] + bias[n]) , optional relu.
// A: bf16 [M x K] row-major (lda), B: bf16 [N x K] row-major (ldb) -> "NT" gemm.
// Block tile: 128 x BN, BK = 32, 256 threads = 8 waves, double-buffered LDS.
template <int BN, int WAVES_N>
__global__ __launch_bounds__(256) void wmma_gemm(
    const u16* __restrict__ A, int lda,
    const u16* __restrict__ Bm, int ldb,
    const float* __restrict__ bias,
    float* __restrict__ Cf, u16* __restrict__ Cb, int ldc,
    int K, float alpha, int doRelu, int transV) {

    constexpr int WAVES_M = 8 / WAVES_N;
    constexpr int WMT = 128 / (16 * WAVES_M);   // M tiles per wave
    constexpr int WNT = BN  / (16 * WAVES_N);   // N tiles per wave
    constexpr int NA  = (128 * 4) / 256;        // uint4 copies per thread for A tile (2)
    constexpr int NB  = (BN  * 4) / 256;        // uint4 copies per thread for B tile (2 or 1)

    const int tid  = threadIdx.x;
    const int lane = tid & 31;
    const int wave = tid >> 5;
    const int wm   = wave / WAVES_N;
    const int wn   = wave % WAVES_N;
    const int waveRow = wm * WMT * 16;
    const int waveCol = wn * WNT * 16;
    const int blockM  = blockIdx.y * 128;
    const int blockN  = blockIdx.x * BN;

    __shared__ uint4 sA[2][128 * 4];   // 128 rows x 32 bf16 per buffer
    __shared__ uint4 sB[2][BN  * 4];

    // per-thread staging source pointers (hoisted out of the K loop)
    const u16* aSrc[NA];
    const u16* bSrc[NB];
#pragma unroll
    for (int j = 0; j < NA; ++j) {
        int i = tid + j * 256, row = i >> 2, q = i & 3;
        aSrc[j] = A + (size_t)(blockM + row) * lda + q * 8;
    }
#pragma unroll
    for (int j = 0; j < NB; ++j) {
        int i = tid + j * 256, row = i >> 2, q = i & 3;
        bSrc[j] = Bm + (size_t)(blockN + row) * ldb + q * 8;
    }

    const int lr = lane & 15;         // row/col within 16
    const int lh = lane >> 4;         // lane-half selects K group

    v8f acc[WMT][WNT];
    const v8f vzero = {0.f, 0.f, 0.f, 0.f, 0.f, 0.f, 0.f, 0.f};
#pragma unroll
    for (int i = 0; i < WMT; ++i)
#pragma unroll
        for (int j = 0; j < WNT; ++j) acc[i][j] = vzero;

    int aoff[WMT], boff[WNT];
#pragma unroll
    for (int tm = 0; tm < WMT; ++tm)
        aoff[tm] = (waveRow + tm * 16 + lr) * 32 + lh * 8;
#pragma unroll
    for (int tn = 0; tn < WNT; ++tn)
        boff[tn] = (waveCol + tn * 16 + lr) * 32 + lh * 8;

    auto compute = [&](int buf) {
        const u16* pA = (const u16*)sA[buf];
        const u16* pB = (const u16*)sB[buf];
        v16bf fa[WMT], fb[WNT];
#pragma unroll
        for (int tm = 0; tm < WMT; ++tm) {
            ((uint4*)&fa[tm])[0] = *(const uint4*)(pA + aoff[tm]);
            ((uint4*)&fa[tm])[1] = *(const uint4*)(pA + aoff[tm] + 16);
        }
#pragma unroll
        for (int tn = 0; tn < WNT; ++tn) {
            ((uint4*)&fb[tn])[0] = *(const uint4*)(pB + boff[tn]);
            ((uint4*)&fb[tn])[1] = *(const uint4*)(pB + boff[tn] + 16);
        }
#pragma unroll
        for (int tm = 0; tm < WMT; ++tm)
#pragma unroll
            for (int tn = 0; tn < WNT; ++tn)
                acc[tm][tn] = __builtin_amdgcn_wmma_f32_16x16x32_bf16(
                    false, fa[tm], false, fb[tn], (short)0, acc[tm][tn], false, false);
    };

#if USE_ASYNC_LDS
    // ---- CDNA5 async global->LDS pipeline (ASYNCcnt tracked) ----
    auto issue = [&](int buf, int k0) {
#pragma unroll
        for (int j = 0; j < NA; ++j)
            async_copy_b128(aSrc[j] + k0, &sA[buf][tid + j * 256]);
#pragma unroll
        for (int j = 0; j < NB; ++j)
            async_copy_b128(bSrc[j] + k0, &sB[buf][tid + j * 256]);
    };

    issue(0, 0);
    int kt = 0;
    for (int k0 = 0; k0 < K; k0 += 32, ++kt) {
        const int cur = kt & 1;
        const bool hasNext = (k0 + 32) < K;
        if (hasNext) {
            issue(cur ^ 1, k0 + 32);     // next tile stays in flight
            wait_async<NA + NB>();       // current tile's copies complete
        } else {
            wait_async<0>();
        }
        __syncthreads();
        compute(cur);
        __syncthreads();
    }
#else
    // ---- fallback: register-staged, software-pipelined copy ----
    uint4 ra[NA], rb[NB];
    auto gload = [&](int k0) {
#pragma unroll
        for (int j = 0; j < NA; ++j) ra[j] = *(const uint4*)(aSrc[j] + k0);
#pragma unroll
        for (int j = 0; j < NB; ++j) rb[j] = *(const uint4*)(bSrc[j] + k0);
    };
    auto sstore = [&](int buf) {
#pragma unroll
        for (int j = 0; j < NA; ++j) sA[buf][tid + j * 256] = ra[j];
#pragma unroll
        for (int j = 0; j < NB; ++j) sB[buf][tid + j * 256] = rb[j];
    };

    gload(0);
    sstore(0);
    __syncthreads();
    int kt = 0;
    for (int k0 = 0; k0 < K; k0 += 32, ++kt) {
        const int cur = kt & 1;
        const bool hasNext = (k0 + 32) < K;
        if (hasNext) {
            gload(k0 + 32);                       // issue global loads early
            __builtin_prefetch(aSrc[0] + k0 + 64, 0, 0);
        }
        compute(cur);                             // overlap with loads in flight
        if (hasNext) sstore(cur ^ 1);             // waits loads, fills other buffer
        __syncthreads();
    }
#endif

    // epilogue
#pragma unroll
    for (int tm = 0; tm < WMT; ++tm) {
#pragma unroll
        for (int tn = 0; tn < WNT; ++tn) {
            const int n = blockN + waveCol + tn * 16 + lr;
            const float bv = bias ? bias[n] : 0.f;
#pragma unroll
            for (int r = 0; r < 8; ++r) {
                const int m = blockM + waveRow + tm * 16 + lh * 8 + r;
                float v = (acc[tm][tn][r] + bv) * alpha;
                if (doRelu) v = fmaxf(v, 0.f);
                if (Cf) Cf[(size_t)m * ldc + n] = v;
                if (Cb) {
                    size_t idx = transV
                        ? ((size_t)((m >> 10) * 1024 + n) * 1024 + (size_t)(m & 1023))
                        : ((size_t)m * ldc + n);
                    Cb[idx] = f2bf(v);
                }
            }
        }
    }
}

// ---------------- masked softmax row kernel ----------------
__global__ __launch_bounds__(256) void softmax_mask_bf16(
    const float* __restrict__ scores, const int* __restrict__ maskRow,
    u16* __restrict__ attn) {
    const int row = blockIdx.x;
    const int tid = threadIdx.x;
    const float* sr = scores + (size_t)row * 1024;
    __shared__ float red[256];

    float mx = -3.0e38f;
    for (int c = tid; c < 1024; c += 256) {
        float v = maskRow[c] ? sr[c] : -1e20f;
        mx = fmaxf(mx, v);
    }
    red[tid] = mx; __syncthreads();
    for (int s = 128; s > 0; s >>= 1) {
        if (tid < s) red[tid] = fmaxf(red[tid], red[tid + s]);
        __syncthreads();
    }
    mx = red[0]; __syncthreads();

    float sum = 0.f;
    for (int c = tid; c < 1024; c += 256) {
        float v = maskRow[c] ? sr[c] : -1e20f;
        sum += __expf(v - mx);
    }
    red[tid] = sum; __syncthreads();
    for (int s = 128; s > 0; s >>= 1) {
        if (tid < s) red[tid] += red[tid + s];
        __syncthreads();
    }
    const float inv = 1.f / red[0];

    u16* ar = attn + (size_t)row * 1024;
    for (int c = tid; c < 1024; c += 256) {
        float v = maskRow[c] ? sr[c] : -1e20f;
        ar[c] = f2bf(__expf(v - mx) * inv);
    }
}

// ---------------- residual + layernorm (+ optional mask, bf16 copy) ----------------
__global__ __launch_bounds__(256) void ln_kernel(
    const float* __restrict__ a, const float* __restrict__ r,
    const float* __restrict__ g, const float* __restrict__ be,
    const int* __restrict__ maskp,
    float* __restrict__ yf, u16* __restrict__ yb) {
    const int row = blockIdx.x;
    const int tid = threadIdx.x;
    const size_t base = (size_t)row * 1024;
    __shared__ float red1[256], red2[256];

    float s = 0.f, s2 = 0.f;
    for (int c = tid; c < 1024; c += 256) {
        float t = a[base + c] + r[base + c];
        s += t; s2 += t * t;
    }
    red1[tid] = s; red2[tid] = s2; __syncthreads();
    for (int st = 128; st > 0; st >>= 1) {
        if (tid < st) { red1[tid] += red1[tid + st]; red2[tid] += red2[tid + st]; }
        __syncthreads();
    }
    const float mu  = red1[0] * (1.f / 1024.f);
    const float var = red2[0] * (1.f / 1024.f) - mu * mu;
    const float rin = rsqrtf(var + 1e-5f);
    const float mval = maskp ? (maskp[row] ? 1.f : 0.f) : 1.f;

    for (int c = tid; c < 1024; c += 256) {
        float t = a[base + c] + r[base + c];
        float y = ((t - mu) * rin * g[c] + be[c]) * mval;
        yf[base + c] = y;
        if (yb) yb[base + c] = f2bf(y);
    }
}

// ---------------- orchestration ----------------
extern "C" void kernel_launch(void* const* d_in, const int* in_sizes, int n_in,
                              void* d_out, int out_size, void* d_ws, size_t ws_size,
                              hipStream_t stream) {
    (void)in_sizes; (void)n_in; (void)out_size; (void)ws_size;
    constexpr int Bc = 4, Sc = 1024, Dc = 1024, Hc = 16, FFc = 4096;
    constexpr size_t BS = (size_t)Bc * Sc;          // 4096 rows

    const float* x    = (const float*)d_in[0];
    const int*   mask = (const int*)  d_in[1];
    const float* Wq = (const float*)d_in[2];  const float* bq = (const float*)d_in[3];
    const float* Wk = (const float*)d_in[4];  const float* bk = (const float*)d_in[5];
    const float* Wv = (const float*)d_in[6];  const float* bv = (const float*)d_in[7];
    const float* Wo = (const float*)d_in[8];  const float* bo = (const float*)d_in[9];
    const float* W1 = (const float*)d_in[10]; const float* b1 = (const float*)d_in[11];
    const float* W2 = (const float*)d_in[12]; const float* b2 = (const float*)d_in[13];
    const float* g1 = (const float*)d_in[14]; const float* be1 = (const float*)d_in[15];
    const float* g2 = (const float*)d_in[16]; const float* be2 = (const float*)d_in[17];
    float* out = (float*)d_out;

    // workspace carve (all 256B aligned)
    char* w = (char*)d_ws;
    auto carve = [&](size_t bytes) -> void* {
        void* p = (void*)w;
        w += (bytes + 255) & ~(size_t)255;
        return p;
    };
    u16*  xb   = (u16*) carve(BS * Dc * 2);                 // x bf16
    u16*  Wqb  = (u16*) carve((size_t)Dc * Dc * 2);
    u16*  Wkb  = (u16*) carve((size_t)Dc * Dc * 2);
    u16*  Wvb  = (u16*) carve((size_t)Dc * Dc * 2);
    u16*  Wob  = (u16*) carve((size_t)Dc * Dc * 2);
    u16*  W1b  = (u16*) carve((size_t)FFc * Dc * 2);
    u16*  W2b  = (u16*) carve((size_t)Dc * FFc * 2);
    u16*  Qb   = (u16*) carve(BS * Dc * 2);                 // Q (already /sqrt(dh)) bf16
    u16*  Kb   = (u16*) carve(BS * Dc * 2);
    u16*  Vt   = (u16*) carve(BS * Dc * 2);                 // V transposed per batch: [B, D, S]
    float* scores = (float*)carve((size_t)Sc * Sc * 4);     // per-(b,h), reused
    u16*  attnb  = (u16*) carve((size_t)Sc * Sc * 2);       // per-(b,h), reused
    u16*  ctx    = (u16*) carve(BS * Dc * 2);
    float* attn_out = (float*)carve(BS * Dc * 4);           // reused as ffn_out later
    float* x1f  = (float*)carve(BS * Dc * 4);
    u16*  x1b   = (u16*) carve(BS * Dc * 2);
    u16*  ffh   = (u16*) carve(BS * (size_t)FFc * 2);
    float* ffn_out = attn_out;                              // alias: attn_out dead after LN1

    // 1) bf16 casts
    cast_f32_bf16<<<2048, 256, 0, stream>>>(x,  xb,  (int)(BS * Dc));
    cast_f32_bf16<<<1024, 256, 0, stream>>>(Wq, Wqb, Dc * Dc);
    cast_f32_bf16<<<1024, 256, 0, stream>>>(Wk, Wkb, Dc * Dc);
    cast_f32_bf16<<<1024, 256, 0, stream>>>(Wv, Wvb, Dc * Dc);
    cast_f32_bf16<<<1024, 256, 0, stream>>>(Wo, Wob, Dc * Dc);
    cast_f32_bf16<<<2048, 256, 0, stream>>>(W1, W1b, FFc * Dc);
    cast_f32_bf16<<<2048, 256, 0, stream>>>(W2, W2b, Dc * FFc);

    // 2) Q/K/V projections (bf16 out). Q pre-scaled by 1/sqrt(64)=0.125.
    dim3 gProj(Dc / 128, (int)(BS / 128));
    wmma_gemm<128, 2><<<gProj, 256, 0, stream>>>(xb, Dc, Wqb, Dc, bq,
        nullptr, Qb, Dc, Dc, 0.125f, 0, 0);
    wmma_gemm<128, 2><<<gProj, 256, 0, stream>>>(xb, Dc, Wkb, Dc, bk,
        nullptr, Kb, Dc, Dc, 1.f, 0, 0);
    wmma_gemm<128, 2><<<gProj, 256, 0, stream>>>(xb, Dc, Wvb, Dc, bv,
        nullptr, Vt, Dc, Dc, 1.f, 0, /*transV=*/1);

    // 3) attention, per (b,h) with reused score/prob scratch
    for (int b = 0; b < Bc; ++b) {
        for (int h = 0; h < Hc; ++h) {
            const u16* Qh  = Qb + (size_t)b * Sc * Dc + h * 64;
            const u16* Kh  = Kb + (size_t)b * Sc * Dc + h * 64;
            const u16* Vth = Vt + ((size_t)b * Dc + (size_t)h * 64) * Sc;
            // scores = Qh @ Kh^T  (M=1024, N=1024, K=64)
            wmma_gemm<128, 2><<<dim3(8, 8), 256, 0, stream>>>(
                Qh, Dc, Kh, Dc, nullptr, scores, nullptr, Sc, 64, 1.f, 0, 0);
            // masked softmax -> bf16 probs
            softmax_mask_bf16<<<Sc, 256, 0, stream>>>(scores, mask + (size_t)b * Sc, attnb);
            // ctx = probs @ V  (M=1024, N=64, K=1024) via transposed V
            wmma_gemm<64, 1><<<dim3(1, 8), 256, 0, stream>>>(
                attnb, Sc, Vth, Sc, nullptr, nullptr,
                ctx + (size_t)b * Sc * Dc + h * 64, Dc, Sc, 1.f, 0, 0);
        }
    }

    // 4) output projection (fp32 out)
    wmma_gemm<128, 2><<<gProj, 256, 0, stream>>>(ctx, Dc, Wob, Dc, bo,
        attn_out, nullptr, Dc, Dc, 1.f, 0, 0);

    // 5) LN1(x + attn_out) -> x1 (fp32 + bf16)
    ln_kernel<<<(int)BS, 256, 0, stream>>>(x, attn_out, g1, be1, nullptr, x1f, x1b);

    // 6) FFN1: relu(x1 @ W1^T + b1) -> bf16 [4096, 4096]
    wmma_gemm<128, 2><<<dim3(FFc / 128, (int)(BS / 128)), 256, 0, stream>>>(
        x1b, Dc, W1b, Dc, b1, nullptr, ffh, FFc, Dc, 1.f, /*relu=*/1, 0);

    // 7) FFN2: ffh @ W2^T + b2 -> fp32
    wmma_gemm<128, 2><<<dim3(Dc / 128, (int)(BS / 128)), 256, 0, stream>>>(
        ffh, FFc, W2b, FFc, b2, ffn_out, nullptr, Dc, FFc, 1.f, 0, 0);

    // 8) LN2(x1 + ffn) * mask -> out
    ln_kernel<<<(int)BS, 256, 0, stream>>>(x1f, ffn_out, g2, be2, mask, out, nullptr);
}
